// ThermalPhysicsPrior_13520557048335
// MI455X (gfx1250) — compile-verified, matrix operand order checked by reading, not run on hardware
//
#include <hip/hip_runtime.h>

#define B_ 4
#define H_ 512
#define W_ 512

// ---------------------------------------------------------------------------
// Types for CDNA5 WMMA (wave32): bf16 16x16x32, f32 accumulate
// ---------------------------------------------------------------------------
typedef __attribute__((ext_vector_type(16))) __bf16 v16bf;
typedef __attribute__((ext_vector_type(8)))  float  v8f;
typedef __attribute__((ext_vector_type(4)))  unsigned int v4u;
typedef __attribute__((ext_vector_type(8)))  int v8i;
typedef __attribute__((ext_vector_type(4)))  int v4i;

#if defined(__has_builtin)
#if __has_builtin(__builtin_amdgcn_tensor_load_to_lds) && \
    __has_builtin(__builtin_amdgcn_s_wait_tensorcnt)
#define USE_TDM 1
#endif
#endif

union Frag16 {            // 16 bf16 = 32 bytes = 8 dwords = 2 int4
    v16bf m;
    int4  q[2];
    int   d[8];
};

__device__ __forceinline__ unsigned f2bf(float f) {
    unsigned u = __float_as_uint(f);
    return (u + 0x7FFFu + ((u >> 16) & 1u)) >> 16;   // round-to-nearest-even
}
__device__ __forceinline__ int pack2bf(float lo, float hi) {
    return (int)(f2bf(lo) | (f2bf(hi) << 16));
}
__device__ __forceinline__ v8f v8f_zero() {
    v8f z = {0.f, 0.f, 0.f, 0.f, 0.f, 0.f, 0.f, 0.f};
    return z;
}

// ---------------------------------------------------------------------------
// 5-tap separable Gaussian (sigma=2), normalized
// ---------------------------------------------------------------------------
__device__ __constant__ float GW5[5] = {
    0.15246915f, 0.22184131f, 0.25137913f, 0.22184131f, 0.15246915f
};

__global__ __launch_bounds__(256) void blur_h(const float* __restrict__ src,
                                              float* __restrict__ dst) {
    int idx = blockIdx.x * 256 + threadIdx.x;          // B*H*W threads
    int x = idx & (W_ - 1);
    int row = idx >> 9;                                // b*H + y
    float s = 0.f;
#pragma unroll
    for (int i = 0; i < 5; ++i) {
        int xx = x + i - 2;
        float v = ((unsigned)xx < W_) ? src[row * W_ + xx] : 0.f;
        s += GW5[i] * v;
    }
    dst[idx] = s;
}

__global__ __launch_bounds__(256) void blur_v(const float* __restrict__ src,
                                              float* __restrict__ dst) {
    int idx = blockIdx.x * 256 + threadIdx.x;
    int x = idx & (W_ - 1);
    int r = idx >> 9;
    int y = r & (H_ - 1);
    int b = r >> 9;
    float s = 0.f;
#pragma unroll
    for (int i = 0; i < 5; ++i) {
        int yy = y + i - 2;
        float v = ((unsigned)yy < H_) ? src[((b * H_ + yy) * W_) + x] : 0.f;
        s += GW5[i] * v;
    }
    dst[idx] = s;
}

// ---------------------------------------------------------------------------
// Sobel magnitude + pack feats = [B][H][W][2] bf16 (ch0 = |grad|, ch1 = blur)
// ---------------------------------------------------------------------------
__global__ __launch_bounds__(256) void feats_k(const float* __restrict__ x,
                                               const float* __restrict__ hblur,
                                               int* __restrict__ feats) {
    int idx = blockIdx.x * 256 + threadIdx.x;
    int px = idx & (W_ - 1);
    int r = idx >> 9;
    int y = r & (H_ - 1);
    int b = r >> 9;

    auto ld = [&](int yy, int xx) -> float {
        if ((unsigned)yy >= H_ || (unsigned)xx >= W_) return 0.f;
        return x[(b * H_ + yy) * W_ + xx];
    };
    float a00 = ld(y - 1, px - 1), a01 = ld(y - 1, px), a02 = ld(y - 1, px + 1);
    float a10 = ld(y,     px - 1),                      a12 = ld(y,     px + 1);
    float a20 = ld(y + 1, px - 1), a21 = ld(y + 1, px), a22 = ld(y + 1, px + 1);
    float gx = (a00 + 2.f * a10 + a20) - (a02 + 2.f * a12 + a22);
    float gy = (a00 + 2.f * a01 + a02) - (a20 + 2.f * a21 + a22);
    float mag = sqrtf(gx * gx + gy * gy);
    feats[idx] = pack2bf(mag, hblur[idx]);
}

// ---------------------------------------------------------------------------
// Weight repack: conv1 (wpack1) + folded BN coefs. A-matrix 16-bit layout:
// lane L -> row M=L%16, half h=L/16; VGPR j: j<4 -> k=h*8+2j, else k=16+h*8+2(j-4)
// conv1 K-order: k = tap*2 + ic (18 real, zero-padded to 32)
// ---------------------------------------------------------------------------
__global__ __launch_bounds__(256) void prep_small(
    const float* __restrict__ w1, const float* __restrict__ b1,
    const float* __restrict__ g1, const float* __restrict__ be1,
    const float* __restrict__ b2, const float* __restrict__ g2,
    const float* __restrict__ be2,
    int* __restrict__ wpack1, float2* __restrict__ coef1,
    float2* __restrict__ coef2) {
    int tid = blockIdx.x * 256 + threadIdx.x;
    const float inv = 0.9999950000374969f;             // 1/sqrt(1+1e-5)
    if (tid < 1024) {                                  // 4 ocg * 32 lanes * 8 dwords
        int j = tid & 7, lane = (tid >> 3) & 31, og = tid >> 8;
        int oc = og * 16 + (lane & 15), h = lane >> 4;
        int pl = (j < 4) ? (h * 4 + j) : (8 + h * 4 + (j - 4));  // k-pair index
        int v = 0;
        if (pl < 9) {
            int ky = pl / 3, kx = pl % 3;
            float lo = w1[((oc * 2 + 0) * 3 + ky) * 3 + kx];
            float hi = w1[((oc * 2 + 1) * 3 + ky) * 3 + kx];
            v = pack2bf(lo, hi);
        }
        wpack1[tid] = v;
    } else if (tid < 1024 + 64) {
        int oc = tid - 1024;
        float gi = g1[oc] * inv;
        coef1[oc] = make_float2(gi, gi * b1[oc] + be1[oc]);
    } else if (tid < 1024 + 64 + 128) {
        int oc = tid - 1088;
        float gi = g2[oc] * inv;
        coef2[oc] = make_float2(gi, gi * b2[oc] + be2[oc]);
    }
}

// conv2 K-order: k = tap*64 + ic  (576 = 18 ksteps of 32)
__global__ __launch_bounds__(256) void repack_w2(const float* __restrict__ w2,
                                                 int* __restrict__ wpack2) {
    int tid = blockIdx.x * 256 + threadIdx.x;          // 36864 threads
    int j = tid & 7, lane = (tid >> 3) & 31, og = (tid >> 8) & 7, ks = tid >> 11;
    int oc = og * 16 + (lane & 15), h = lane >> 4;
    int pl = (j < 4) ? (h * 4 + j) : (8 + h * 4 + (j - 4));
    int k0 = ks * 32 + 2 * pl;
    int tap = k0 >> 6, ic = k0 & 63;
    int ky = tap / 3, kx = tap % 3;
    float lo = w2[((oc * 64 + ic) * 3 + ky) * 3 + kx];
    float hi = w2[((oc * 64 + ic + 1) * 3 + ky) * 3 + kx];
    wpack2[tid] = pack2bf(lo, hi);
}

// ---------------------------------------------------------------------------
// Conv1: feats[B,H,W,2]bf16 -> y1[B,H,W,64]bf16 via WMMA, BN+ReLU fused.
// Block = 1 row (8 waves * 64px). Wave: 4 oc-groups x 4 px-groups, 1 kstep.
// ---------------------------------------------------------------------------
__global__ __launch_bounds__(256) void conv1_wmma(
    const int* __restrict__ feats, const int4* __restrict__ wpack1,
    const float2* __restrict__ coef1, int4* __restrict__ y1) {
    const int tid = threadIdx.x;
    const int wave = tid >> 5, lane = tid & 31;
    const int ln = lane & 15, h = lane >> 4;
    const int row = blockIdx.x;                        // b*H + y
    const int b = row >> 9, y = row & (H_ - 1);
    const int x0 = wave * 64;

    // B fragments: 4 pixel groups of 16
    Frag16 bf[4];
#pragma unroll
    for (int pg = 0; pg < 4; ++pg) {
        int px = x0 + pg * 16 + ln;
#pragma unroll
        for (int j = 0; j < 8; ++j) {
            int pl = (j < 4) ? (h * 4 + j) : (8 + h * 4 + (j - 4));
            int v = 0;
            if (pl < 9) {
                int yy = y + pl / 3 - 1;
                int xx = px + pl % 3 - 1;
                if ((unsigned)yy < H_ && (unsigned)xx < W_)
                    v = feats[(b * H_ + yy) * W_ + xx];
            }
            bf[pg].d[j] = v;
        }
    }

    v8f acc[4][4];
#pragma unroll
    for (int og = 0; og < 4; ++og) {
        Frag16 a;
        const int4* wp = wpack1 + (og * 32 + lane) * 2;
        a.q[0] = wp[0];
        a.q[1] = wp[1];
#pragma unroll
        for (int pg = 0; pg < 4; ++pg)
            acc[og][pg] = __builtin_amdgcn_wmma_f32_16x16x32_bf16(
                false, a.m, false, bf[pg].m, (short)0, v8f_zero(), false, false);
    }

    // Epilogue: BN+ReLU, pack 8 consecutive oc (= v index) as bf16, b128 store
#pragma unroll
    for (int og = 0; og < 4; ++og) {
#pragma unroll
        for (int pg = 0; pg < 4; ++pg) {
            int px = x0 + pg * 16 + ln;
            int od[4];
#pragma unroll
            for (int p = 0; p < 4; ++p) {
                int oc0 = og * 16 + 8 * h + 2 * p;
                float2 c0 = coef1[oc0], c1 = coef1[oc0 + 1];
                float v0 = fmaxf(acc[og][pg][2 * p] * c0.x + c0.y, 0.f);
                float v1 = fmaxf(acc[og][pg][2 * p + 1] * c1.x + c1.y, 0.f);
                od[p] = pack2bf(v0, v1);
            }
            int4 st = make_int4(od[0], od[1], od[2], od[3]);
            // y1 int4 index: pixel*8 + (og*16+8h)/8
            y1[(((b * H_ + y) * W_ + px) << 3) + og * 2 + h] = st;
        }
    }
}

// ---------------------------------------------------------------------------
// Conv2: y1[B,H,W,64]bf16 -> out[B,128,H,W]f32.  Weights staged in LDS
// (147456 B dynamic) by the Tensor Data Mover (1D D#: 18432 x 8B elements);
// fallback is a cooperative vector-load loop.  Block = 1 row.  Wave:
// 8 oc-groups x 4 px-groups, 18 ksteps -> 576 v_wmma per wave,
// 8x global_load_b128 (B) per kstep, nontemporal f32 stores.
// ---------------------------------------------------------------------------
__global__ __launch_bounds__(256) void conv2_wmma(
    const int4* __restrict__ y1, const int4* __restrict__ wpack2,
    const float2* __restrict__ coef2, const float* __restrict__ scale_p,
    float* __restrict__ out) {
    extern __shared__ int4 smem[];                     // 9216 int4 = 147456 B

    const int tid = threadIdx.x;
    const int wave = tid >> 5, lane = tid & 31;
    const int ln = lane & 15, h = lane >> 4;
    const int row = blockIdx.x;                        // b*H + y
    const int b = row >> 9, y = row & (H_ - 1);
    const int x0 = wave * 64;

#if USE_TDM
    if (wave == 0) {
        // Tensor DMA descriptor (D#), groups 0/1; groups 2/3 zero (1D tile).
        unsigned lds_base = (unsigned)(uintptr_t)(void*)smem;
        unsigned long long ga = (unsigned long long)(uintptr_t)wpack2;
        const unsigned ELEMS = 18432u;                 // 147456 B / 8 B
        v4u g0;
        g0.x = 1u;                                     // count=1, user D#
        g0.y = lds_base;                               // lds_addr
        g0.z = (unsigned)(ga & 0xFFFFFFFFull);         // global_addr[31:0]
        g0.w = (unsigned)((ga >> 32) & 0x1FFFFFFull) | 0x80000000u; // [56:32]|type=2
        v8i g1;
        g1[0] = (int)(3u << 16);                       // data_size = 8 B
        g1[1] = (int)((ELEMS & 0xFFFFu) << 16);        // tensor_dim0[15:0]
        g1[2] = (int)(((ELEMS >> 16) & 0xFFFFu) | (1u << 16)); // dim0 hi | dim1=1
        g1[3] = (int)((ELEMS & 0xFFFFu) << 16);        // tile_dim0
        g1[4] = 1;                                     // tile_dim1 = 1
        g1[5] = (int)ELEMS;                            // tensor_dim0_stride lo
        g1[6] = 0;
        g1[7] = 0;
        v4i gz = {0, 0, 0, 0};
        v8i gz8 = {0, 0, 0, 0, 0, 0, 0, 0};
        __builtin_amdgcn_tensor_load_to_lds(g0, g1, gz, gz, gz8, 0);
        __builtin_amdgcn_s_wait_tensorcnt(0);
    }
#else
    for (int i = tid; i < 9216; i += 256) smem[i] = wpack2[i];
#endif
    __syncthreads();

    // Prefetch the three y1 rows this wave will read (L2-resident activations)
#pragma unroll
    for (int dy = -1; dy <= 1; ++dy) {
        int yy = y + dy;
        if ((unsigned)yy < H_) {
            const char* p = (const char*)(y1 + (((b * H_ + yy) * W_ + x0) << 3));
            __builtin_prefetch(p + lane * 256, 0, 0);
        }
    }

    const float sc = *scale_p;

    v8f acc[8][4];
#pragma unroll
    for (int og = 0; og < 8; ++og)
#pragma unroll
        for (int pg = 0; pg < 4; ++pg) acc[og][pg] = v8f_zero();

    const int4 zq = make_int4(0, 0, 0, 0);

    for (int ks = 0; ks < 18; ++ks) {
        const int tap = ks >> 1;
        const int yy = y + tap / 3 - 1;
        const int kx = tap % 3 - 1;
        const bool yok = (unsigned)yy < H_;
        const int icq = ((ks & 1) << 2) + h;           // int4 offset within pixel's 64ch

        Frag16 bf[4];
#pragma unroll
        for (int pg = 0; pg < 4; ++pg) {
            int xx = x0 + pg * 16 + ln + kx;
            if (yok && (unsigned)xx < W_) {
                const int4* p = y1 + (((b * H_ + yy) * W_ + xx) << 3) + icq;
                bf[pg].q[0] = p[0];
                bf[pg].q[1] = p[2];                    // +16 channels
            } else {
                bf[pg].q[0] = zq;
                bf[pg].q[1] = zq;
            }
        }

#pragma unroll
        for (int og = 0; og < 8; ++og) {
            Frag16 a;
            const int4* wp = smem + ((ks * 8 + og) * 32 + lane) * 2;
            a.q[0] = wp[0];
            a.q[1] = wp[1];
#pragma unroll
            for (int pg = 0; pg < 4; ++pg)
                acc[og][pg] = __builtin_amdgcn_wmma_f32_16x16x32_bf16(
                    false, a.m, false, bf[pg].m, (short)0, acc[og][pg], false, false);
        }
    }

    // Epilogue: BN + ReLU + *scale, coalesced nontemporal f32 stores to NCHW
#pragma unroll
    for (int og = 0; og < 8; ++og) {
#pragma unroll
        for (int v = 0; v < 8; ++v) {
            int oc = og * 16 + 8 * h + v;
            float2 c = coef2[oc];
            float* op = out + ((b * 128 + oc) * H_ + y) * W_ + x0 + ln;
#pragma unroll
            for (int pg = 0; pg < 4; ++pg) {
                float val = fmaxf(acc[og][pg][v] * c.x + c.y, 0.f) * sc;
                __builtin_nontemporal_store(val, op + pg * 16);
            }
        }
    }
}

// ---------------------------------------------------------------------------
// Launch
// ---------------------------------------------------------------------------
extern "C" void kernel_launch(void* const* d_in, const int* in_sizes, int n_in,
                              void* d_out, int out_size, void* d_ws, size_t ws_size,
                              hipStream_t stream) {
    const float* x   = (const float*)d_in[0];
    const float* w1  = (const float*)d_in[1];
    const float* b1  = (const float*)d_in[2];
    const float* g1  = (const float*)d_in[3];
    const float* be1 = (const float*)d_in[4];
    const float* w2  = (const float*)d_in[5];
    const float* b2  = (const float*)d_in[6];
    const float* g2  = (const float*)d_in[7];
    const float* be2 = (const float*)d_in[8];
    const float* scl = (const float*)d_in[9];
    float* out = (float*)d_out;

    char* ws = (char*)d_ws;
    const size_t NPIX = (size_t)B_ * H_ * W_;          // 1,048,576
    float* blurA  = (float*)(ws);                      // 4 MB
    float* blurB  = (float*)(ws + 4194304);            // 4 MB
    int*   feats  = (int*)  (ws + 8388608);            // 4 MB (NHWC2 bf16)
    int*   wpack1 = (int*)  (ws + 12582912);           // 4 KB
    float2* coef1 = (float2*)(ws + 12587008);          // 512 B
    float2* coef2 = (float2*)(ws + 12587520);          // 1 KB
    int*   wpack2 = (int*)  (ws + 12648448);           // 147456 B
    int4*  y1     = (int4*) (ws + 13631488);           // 128 MB (NHWC64 bf16)
    (void)in_sizes; (void)n_in; (void)out_size; (void)ws_size; (void)NPIX;

    // Weight repack + BN folding
    prep_small<<<5, 256, 0, stream>>>(w1, b1, g1, be1, b2, g2, be2,
                                      wpack1, coef1, coef2);
    repack_w2<<<144, 256, 0, stream>>>(w2, wpack2);

    // 5 iterations of separable 5x5 Gaussian blur (zero-padded per pass)
    const int PB = (int)(NPIX / 256);                  // 4096
    blur_h<<<PB, 256, 0, stream>>>(x, blurA);
    blur_v<<<PB, 256, 0, stream>>>(blurA, blurB);
    for (int i = 1; i < 5; ++i) {
        blur_h<<<PB, 256, 0, stream>>>(blurB, blurA);
        blur_v<<<PB, 256, 0, stream>>>(blurA, blurB);
    }

    // Sobel magnitude + pack bf16 feature map
    feats_k<<<PB, 256, 0, stream>>>(x, blurB, feats);

    // conv-bn-relu x2 via WMMA implicit GEMM
    conv1_wmma<<<B_ * H_, 256, 0, stream>>>(feats, (const int4*)wpack1, coef1, y1);
    conv2_wmma<<<B_ * H_, 256, 147456, stream>>>(y1, (const int4*)wpack2, coef2,
                                                 scl, out);
}
